// PointerDecoder_67748814127759
// MI455X (gfx1250) — compile-verified, matrix operand order checked by reading, not run on hardware
//
#include <hip/hip_runtime.h>
#include <hip/hip_bf16.h>

// ---------------------------------------------------------------------------
// PointerDecoder fused decoder for gfx1250 (MI455X).
// B=128, N=2048, D=512. Core GEMM (H @ Wk^T, 137 GFLOP) runs on
// v_wmma_f32_16x16x32_bf16 with f32 accumulation, fused with the
// tanh(Q+K).v -> 10*tanh epilogue so K (536 MB) is never materialized.
// 64-row n-tiles per block halve Bfrag L2 traffic vs 32-row tiles.
// ---------------------------------------------------------------------------

#define BB 128
#define NN 2048
#define DD 512

typedef __attribute__((ext_vector_type(16))) __bf16         v16bf;
typedef __attribute__((ext_vector_type(8)))  float          v8f;
typedef __attribute__((ext_vector_type(8)))  unsigned short ushort8;

static __device__ __forceinline__ unsigned short f32_to_bf16_bits(float f) {
    unsigned int u = __builtin_bit_cast(unsigned int, f);
    // round to nearest even
    unsigned int r = u + 0x7FFFu + ((u >> 16) & 1u);
    return (unsigned short)(r >> 16);
}

// Hardware transcendental tanh if the toolchain exposes it (V_TANH_F32).
static __device__ __forceinline__ float fast_tanh(float x) {
#if __has_builtin(__builtin_amdgcn_tanhf)
    return __builtin_amdgcn_tanhf(x);
#elif __has_builtin(__builtin_amdgcn_tanh_f32)
    return __builtin_amdgcn_tanh_f32(x);
#else
    return tanhf(x);
#endif
}

// ---------------------------------------------------------------------------
// 1) graph_emb[b][d] = mean_n H[b][n][d]
//    128 blocks x 1024 threads; thread = (n-chunk, d-quad); float4 loads.
// ---------------------------------------------------------------------------
__global__ __launch_bounds__(1024) void pd_mean_kernel(
    const float* __restrict__ H, float* __restrict__ gemb)
{
    __shared__ float red[8 * DD];  // [nchunk][d]
    const int b  = blockIdx.x;
    const int t  = threadIdx.x;
    const int dq = t & 127;   // d-quad index: d = 4*dq .. 4*dq+3
    const int nc = t >> 7;    // n-chunk 0..7 (256 rows each)

    const float* base = H + ((size_t)b * NN + (size_t)nc * 256) * DD + dq * 4;
    float4 s = make_float4(0.f, 0.f, 0.f, 0.f);
    for (int n = 0; n < 256; ++n) {
        float4 hv = *(const float4*)(base + (size_t)n * DD);
        s.x += hv.x; s.y += hv.y; s.z += hv.z; s.w += hv.w;
    }
    *(float4*)&red[nc * DD + dq * 4] = s;
    __syncthreads();
    if (t < DD) {
        float acc = 0.f;
        #pragma unroll
        for (int c = 0; c < 8; ++c) acc += red[c * DD + t];
        gemb[b * DD + t] = acc * (1.0f / (float)NN);
    }
}

// ---------------------------------------------------------------------------
// 2) Pack Wk (f32, row-major [e][d]) into bf16 B-fragments in the exact
//    v_wmma_f32_16x16x32_bf16 B-operand lane layout:
//    Bfrag[dstep][etile][lane][i], i=0..15 contiguous per lane (32 B).
//    Lane mapping (ISA 7.12.2): lanes 0-15 -> K 0..7 (i<8), 16..23 (i>=8);
//    lanes 16-31 -> K 8..15, 24..31. Column n = lane & 15.
// ---------------------------------------------------------------------------
__global__ __launch_bounds__(256) void pd_pack_wk_kernel(
    const float* __restrict__ Wk, unsigned short* __restrict__ Bfrag)
{
    const int idx   = blockIdx.x * 256 + threadIdx.x;  // 0 .. 262143
    const int i     = idx & 15;
    const int lane  = (idx >> 4) & 31;
    const int etile = (idx >> 9) & 31;
    const int dstep = idx >> 14;                       // 0..15
    const int half  = lane >> 4;
    const int k     = (i < 8) ? (i + half * 8) : (16 + (i - 8) + half * 8);
    const int e     = etile * 16 + (lane & 15);
    const int d     = dstep * 32 + k;
    Bfrag[idx] = f32_to_bf16_bits(Wk[(size_t)e * DD + d]);
}

// ---------------------------------------------------------------------------
// 3) q[b][i] = bc[i] + sum_j ctx[b][j] * Wc[i][j], ctx = [gemb, last, cap]
// ---------------------------------------------------------------------------
__global__ __launch_bounds__(256) void pd_ctx_q_kernel(
    const float* __restrict__ gemb, const float* __restrict__ last_emb,
    const float* __restrict__ cap,  const float* __restrict__ Wc,
    const float* __restrict__ bc,   float* __restrict__ qbuf)
{
    const int idx = blockIdx.x * 256 + threadIdx.x;  // b*512 + i
    const int b = idx >> 9, i = idx & (DD - 1);
    const float* wr = Wc + (size_t)i * (2 * DD + 1);
    const float* ge = gemb     + b * DD;
    const float* le = last_emb + b * DD;
    float s = bc[i] + cap[b] * wr[2 * DD];
    for (int j = 0; j < DD; ++j) s += ge[j] * wr[j];
    for (int j = 0; j < DD; ++j) s += le[j] * wr[DD + j];
    qbuf[idx] = s;
}

// ---------------------------------------------------------------------------
// 4) Q[b][e] = sum_i q[b][i] * Wq[e][i]
// ---------------------------------------------------------------------------
__global__ __launch_bounds__(256) void pd_qproj_kernel(
    const float* __restrict__ qbuf, const float* __restrict__ Wq,
    float* __restrict__ Qbuf)
{
    const int idx = blockIdx.x * 256 + threadIdx.x;  // b*512 + e
    const int b = idx >> 9, e = idx & (DD - 1);
    const float* wr = Wq + (size_t)e * DD;
    const float* qq = qbuf + b * DD;
    float s = 0.f;
    for (int j = 0; j < DD; ++j) s += qq[j] * wr[j];
    Qbuf[idx] = s;
}

// ---------------------------------------------------------------------------
// 5) Main fused kernel: scores[b][n] = 10*tanh( sum_e tanh(Q[b][e]+K[b][n][e])
//    * v[e] ), K = H @ Wk^T computed on WMMA bf16, never stored.
//    Block = (b, 64-row n-tile), 8 waves; wave w owns e in [64w, 64w+64).
//    K-loop: d in steps of 32; 64x32 H chunk staged to LDS in A-fragment
//    order by all 256 threads (one ds_store_b128 each, no divergence).
//    16 WMMAs per wave per d-step against 2 global b128 loads per e-tile.
// ---------------------------------------------------------------------------
__global__ __launch_bounds__(256) void pd_scores_kernel(
    const float* __restrict__ H, const unsigned short* __restrict__ Bfrag,
    const float* __restrict__ Qbuf, const float* __restrict__ v,
    const unsigned char* __restrict__ mask, float* __restrict__ out)
{
    __shared__ __align__(16) unsigned short Abuf[4 * 512];  // 4 A-tiles (16x32 bf16)
    __shared__ float partial[64];

    const int t    = threadIdx.x;
    const int lane = t & 31;
    const int wave = t >> 5;
    const int b    = blockIdx.y;
    const int n0   = blockIdx.x * 64;

    if (t < 64) partial[t] = 0.f;

    // Per-lane Q and v for this wave's 4 e-tiles (lane column = lane & 15).
    float q4[4], vv[4];
    #pragma unroll
    for (int et = 0; et < 4; ++et) {
        const int e = wave * 64 + et * 16 + (lane & 15);
        q4[et] = Qbuf[b * DD + e];
        vv[et] = v[e];
    }

    v8f acc[4][4];
    const v8f vzero = {0.f, 0.f, 0.f, 0.f, 0.f, 0.f, 0.f, 0.f};
    #pragma unroll
    for (int nt = 0; nt < 4; ++nt)
        #pragma unroll
        for (int et = 0; et < 4; ++et) acc[nt][et] = vzero;

    // Staging roles: all 256 threads convert the 64x32 f32 H chunk to bf16 in
    // A-fragment order. Thread = (row r, k-group g of 8). One ds_store_b128.
    const int r = t & 63;                 // H row within the 64-row tile
    const int g = t >> 6;                 // k-group 0..3
    const int sm      = r & 15;
    const int sntile  = r >> 4;
    const int slane   = sm + ((g & 1) << 4);   // K-halves alternate lane halves
    const int soff    = sntile * 512 + slane * 16 + ((g >> 1) << 3);
    const float* Hrow = H + ((size_t)b * NN + n0 + r) * DD + g * 8;

    for (int dstep = 0; dstep < 16; ++dstep) {
        {
            const float* src = Hrow + dstep * 32;
            const float4 f0 = *(const float4*)(src);
            const float4 f1 = *(const float4*)(src + 4);
            ushort8 hv;
            hv[0] = f32_to_bf16_bits(f0.x); hv[1] = f32_to_bf16_bits(f0.y);
            hv[2] = f32_to_bf16_bits(f0.z); hv[3] = f32_to_bf16_bits(f0.w);
            hv[4] = f32_to_bf16_bits(f1.x); hv[5] = f32_to_bf16_bits(f1.y);
            hv[6] = f32_to_bf16_bits(f1.z); hv[7] = f32_to_bf16_bits(f1.w);
            *(ushort8*)&Abuf[soff] = hv;
        }
        __syncthreads();

        const v16bf a0 = *(const v16bf*)&Abuf[0 * 512 + lane * 16];
        const v16bf a1 = *(const v16bf*)&Abuf[1 * 512 + lane * 16];
        const v16bf a2 = *(const v16bf*)&Abuf[2 * 512 + lane * 16];
        const v16bf a3 = *(const v16bf*)&Abuf[3 * 512 + lane * 16];
        #pragma unroll
        for (int et = 0; et < 4; ++et) {
            const v16bf bmat = *(const v16bf*)(
                Bfrag + (((size_t)(dstep * 32 + (wave * 4 + et)) * 32 + lane) * 16));
            acc[0][et] = __builtin_amdgcn_wmma_f32_16x16x32_bf16(
                false, a0, false, bmat, (short)0, acc[0][et], false, false);
            acc[1][et] = __builtin_amdgcn_wmma_f32_16x16x32_bf16(
                false, a1, false, bmat, (short)0, acc[1][et], false, false);
            acc[2][et] = __builtin_amdgcn_wmma_f32_16x16x32_bf16(
                false, a2, false, bmat, (short)0, acc[2][et], false, false);
            acc[3][et] = __builtin_amdgcn_wmma_f32_16x16x32_bf16(
                false, a3, false, bmat, (short)0, acc[3][et], false, false);
        }
        __syncthreads();
    }

    // Epilogue. C/D layout: lane column = lane&15; VGPR rr -> row rr + 8*(lane>=16).
    #pragma unroll
    for (int nt = 0; nt < 4; ++nt) {
        #pragma unroll
        for (int rr = 0; rr < 8; ++rr) {
            float s = 0.f;
            #pragma unroll
            for (int et = 0; et < 4; ++et)
                s += fast_tanh(q4[et] + acc[nt][et][rr]) * vv[et];
            // reduce the 16 e-columns (lanes sharing this row) of this wave
            #pragma unroll
            for (int off = 1; off < 16; off <<= 1)
                s += __shfl_xor(s, off, 32);
            if ((lane & 15) == 0) {
                const int brow = nt * 16 + rr + ((lane >> 4) << 3);
                atomicAdd(&partial[brow], s);  // combine waves' e-slices
            }
        }
    }
    __syncthreads();

    if (t < 64) {
        const int n = n0 + t;
        float sc = 10.0f * fast_tanh(partial[t]);
        if (!mask[(size_t)b * NN + n]) sc = -1e9f;
        out[(size_t)b * NN + n] = sc;
    }
}

// ---------------------------------------------------------------------------
// 6) In-place row-wise log_softmax over N=2048. One block per b.
// ---------------------------------------------------------------------------
__global__ __launch_bounds__(256) void pd_logsoftmax_kernel(float* __restrict__ out)
{
    __shared__ float red[256];
    const int b = blockIdx.x, t = threadIdx.x;
    float* row = out + (size_t)b * NN;

    float x[8];
    float m = -3.4e38f;
    #pragma unroll
    for (int i = 0; i < 8; ++i) { x[i] = row[t + i * 256]; m = fmaxf(m, x[i]); }
    red[t] = m;
    __syncthreads();
    for (int s = 128; s > 0; s >>= 1) {
        if (t < s) red[t] = fmaxf(red[t], red[t + s]);
        __syncthreads();
    }
    m = red[0];
    __syncthreads();

    float se = 0.f;
    #pragma unroll
    for (int i = 0; i < 8; ++i) se += expf(x[i] - m);
    red[t] = se;
    __syncthreads();
    for (int s = 128; s > 0; s >>= 1) {
        if (t < s) red[t] += red[t + s];
        __syncthreads();
    }
    const float lse = m + logf(red[0]);
    #pragma unroll
    for (int i = 0; i < 8; ++i) row[t + i * 256] = x[i] - lse;
}

// ---------------------------------------------------------------------------
extern "C" void kernel_launch(void* const* d_in, const int* in_sizes, int n_in,
                              void* d_out, int out_size, void* d_ws, size_t ws_size,
                              hipStream_t stream)
{
    const float*         H        = (const float*)d_in[0];
    const float*         last_emb = (const float*)d_in[1];
    const float*         cap      = (const float*)d_in[2];
    const unsigned char* mask     = (const unsigned char*)d_in[3];
    const float*         Wc       = (const float*)d_in[4];
    const float*         bc       = (const float*)d_in[5];
    const float*         Wq       = (const float*)d_in[6];
    const float*         Wk       = (const float*)d_in[7];
    const float*         v        = (const float*)d_in[8];
    float*               out      = (float*)d_out;

    float* ws_f = (float*)d_ws;
    float*          gemb  = ws_f;                                  // 65536 f32
    float*          qbuf  = ws_f + BB * DD;                        // 65536 f32
    float*          Qbuf  = ws_f + 2 * BB * DD;                    // 65536 f32
    unsigned short* Bfrag = (unsigned short*)(ws_f + 3 * BB * DD); // 262144 u16

    pd_mean_kernel   <<<BB,   1024, 0, stream>>>(H, gemb);
    pd_pack_wk_kernel<<<1024, 256,  0, stream>>>(Wk, Bfrag);
    pd_ctx_q_kernel  <<<BB * DD / 256, 256, 0, stream>>>(gemb, last_emb, cap, Wc, bc, qbuf);
    pd_qproj_kernel  <<<BB * DD / 256, 256, 0, stream>>>(qbuf, Wq, Qbuf);

    dim3 grid(NN / 64, BB);
    pd_scores_kernel <<<grid, 256, 0, stream>>>(H, Bfrag, Qbuf, v, mask, out);
    pd_logsoftmax_kernel<<<BB, 256, 0, stream>>>(out);
}
